// efficient_Attention_modified_21071109554259
// MI455X (gfx1250) — matrix operand order repaired and checked by reading, not compiled, not run on hardware
//
#include <hip/hip_runtime.h>
#include <hip/hip_bf16.h>

// ---------------------------------------------------------------------------
// Efficient (linear) attention on MI455X / gfx1250.
// All einsums go through v_wmma_f32_16x16x32_f16 (wave32 WMMA), with
// register-level software pipelining of the K loops so WMMAs overlap with
// in-flight global_load_b128s instead of draining loadcnt to 0 each step.
// Dims: B=8, DIM=384, N=4096, HEADS=8, KD=32, D=64, NH_KD=256, DH=512.
// ---------------------------------------------------------------------------

#define BB      8
#define DIMC    384
#define NN      4096
#define HEADS   8
#define KDIM    32
#define DD      64
#define NHKD    256
#define DHC     512
#define KSPLIT  8              // K-split factor for the context GEMM
#define NSPL    (NN / KSPLIT)  // 512 spatial positions per split

typedef __attribute__((ext_vector_type(16))) _Float16 v16h;
typedef __attribute__((ext_vector_type(8)))  _Float16 v8h;
typedef __attribute__((ext_vector_type(8)))  float    v8f;

// Build the A-operand v16h: lane holds row (lane&15),
// K = kb..kb+15 where kb = (lane>>4)*16.  Two 16B loads,
// both 16-byte aligned by construction.
__device__ __forceinline__ v16h load_a16(const _Float16* p) {
    v8h lo = *(const v8h*)(p);
    v8h hi = *(const v8h*)(p + 8);
    v16h r;
#pragma unroll
    for (int i = 0; i < 8; ++i) { r[i] = lo[i]; r[i + 8] = hi[i]; }
    return r;
}

__device__ __forceinline__ v8f wmma_f16(v16h a, v16h b, v8f c) {
    return __builtin_amdgcn_wmma_f32_16x16x32_f16(
        /*neg_a=*/false, a, /*neg_b=*/false, b,
        /*c_mod=*/(short)0, c, /*reuse_a=*/false, /*reuse_b=*/false);
}

// ---------------------------------------------------------------------------
// Kernel 0: fold weights to f16, concat Q/K/V weights + BN affine params.
// ---------------------------------------------------------------------------
__global__ void prep_weights_kernel(
    const float* __restrict__ Wq, const float* __restrict__ sq, const float* __restrict__ bq,
    const float* __restrict__ Wk, const float* __restrict__ sk, const float* __restrict__ bk,
    const float* __restrict__ Wv, const float* __restrict__ sv, const float* __restrict__ bv,
    const float* __restrict__ Wp,
    _Float16* __restrict__ wabc, _Float16* __restrict__ wph,
    float* __restrict__ sconc, float* __restrict__ bconc)
{
    const int QK = NHKD * DIMC;            // 98304
    const int VS = DHC * DIMC;             // 196608
    int i = blockIdx.x * blockDim.x + threadIdx.x;
    if (i < QK)               wabc[i] = (_Float16)Wq[i];
    else if (i < 2 * QK)      wabc[i] = (_Float16)Wk[i - QK];
    else if (i < 2 * QK + VS) wabc[i] = (_Float16)Wv[i - 2 * QK];
    if (i < DIMC * DHC)       wph[i]  = (_Float16)Wp[i];
    if (i < NHKD) {
        sconc[i] = sq[i];           bconc[i] = bq[i];
        sconc[NHKD + i] = sk[i];    bconc[NHKD + i] = bk[i];
    }
    if (i < DHC) {
        sconc[2 * NHKD + i] = sv[i];
        bconc[2 * NHKD + i] = bv[i];
    }
}

// ---------------------------------------------------------------------------
// Kernel 1: transpose-convert x[b][c][n] f32 -> xt[b][n][c] f16 (LDS tiled).
// ---------------------------------------------------------------------------
__global__ void xpose_kernel(const float* __restrict__ x, _Float16* __restrict__ xt)
{
    __shared__ float tile[32][33];
    int b = blockIdx.z;
    int nb = blockIdx.x * 32, cb = blockIdx.y * 32;
    int tx = threadIdx.x, ty = threadIdx.y;
    const float* xb = x + (size_t)b * DIMC * NN;
#pragma unroll
    for (int j = 0; j < 32; j += 8)
        tile[ty + j][tx] = xb[(size_t)(cb + ty + j) * NN + nb + tx];
    __syncthreads();
    _Float16* xo = xt + (size_t)b * NN * DIMC;
#pragma unroll
    for (int j = 0; j < 32; j += 8)
        xo[(size_t)(nb + ty + j) * DIMC + cb + tx] = (_Float16)tile[tx][ty + j];
}

// ---------------------------------------------------------------------------
// Kernel 2: fused QKV projection GEMM + BN affine (software pipelined).
// A = wabc [1024][384] row-major, B = xt[b][n][384] (K-contiguous per column).
// Block: 256 threads = 8 waves; wave tile 16x64; block tile 64x128.
// ---------------------------------------------------------------------------
__global__ void gemm_qkv_kernel(const _Float16* __restrict__ A,
                                const _Float16* __restrict__ Bt,
                                const float* __restrict__ scale,
                                const float* __restrict__ bias,
                                _Float16* __restrict__ qt,   // [B][N][256]
                                _Float16* __restrict__ kh,   // [B][256][N]
                                _Float16* __restrict__ vh)   // [B][512][N]
{
    const int K = DIMC;
    int b    = blockIdx.z;
    int lane = threadIdx.x & 31;
    int wave = threadIdx.x >> 5;
    int mw = wave >> 1, nw = wave & 1;
    int Mbase = blockIdx.y * 64 + mw * 16;
    int Nbase = blockIdx.x * 128 + nw * 64;
    int row  = lane & 15;
    int half = lane >> 4;
    const _Float16* Ar = A + (size_t)(Mbase + row) * K + half * 16;
    const _Float16* Bb = Bt + ((size_t)b * NN + Nbase) * K + half * 16;

    v8f acc[4] = {};
    // prologue: stage 0 fragments
    v16h a0 = load_a16(Ar);
    v16h b0[4];
#pragma unroll
    for (int ns = 0; ns < 4; ++ns)
        b0[ns] = *(const v16h*)(Bb + (size_t)(ns * 16 + row) * K);
    // pipelined main loop: prefetch k+1 while multiplying k
    for (int kc = 0; kc < K - 32; kc += 32) {
        v16h a1 = load_a16(Ar + kc + 32);
        v16h b1[4];
#pragma unroll
        for (int ns = 0; ns < 4; ++ns)
            b1[ns] = *(const v16h*)(Bb + (size_t)(ns * 16 + row) * K + kc + 32);
#pragma unroll
        for (int ns = 0; ns < 4; ++ns)
            acc[ns] = wmma_f16(a0, b0[ns], acc[ns]);
        a0 = a1;
#pragma unroll
        for (int ns = 0; ns < 4; ++ns) b0[ns] = b1[ns];
    }
#pragma unroll
    for (int ns = 0; ns < 4; ++ns)
        acc[ns] = wmma_f16(a0, b0[ns], acc[ns]);

#pragma unroll
    for (int ns = 0; ns < 4; ++ns) {
#pragma unroll
        for (int r = 0; r < 8; ++r) {
            int o = Mbase + r + half * 8;
            int n = Nbase + ns * 16 + row;
            _Float16 h = (_Float16)(acc[ns][r] * scale[o] + bias[o]);
            if (Mbase < NHKD)           // Q -> transposed [b][n][256]
                qt[((size_t)b * NN + n) * NHKD + o] = h;
            else if (Mbase < 2 * NHKD)  // K -> [b][256][N]
                kh[((size_t)b * NHKD + (o - NHKD)) * NN + n] = h;
            else                        // V -> [b][512][N]
                vh[((size_t)b * DHC + (o - 2 * NHKD)) * NN + n] = h;
        }
    }
}

// ---------------------------------------------------------------------------
// Kernel 3: softmax of K over the spatial dim (rows of length 4096), in-place.
// ---------------------------------------------------------------------------
__global__ void softmax_k_kernel(_Float16* __restrict__ kh)
{
    __shared__ float red[256];
    _Float16* p = kh + (size_t)blockIdx.x * NN;
    int t = threadIdx.x;
    float mx = -1e30f;
    for (int i = t; i < NN; i += 256) mx = fmaxf(mx, (float)p[i]);
    red[t] = mx; __syncthreads();
    for (int s = 128; s > 0; s >>= 1) {
        if (t < s) red[t] = fmaxf(red[t], red[t + s]);
        __syncthreads();
    }
    mx = red[0]; __syncthreads();
    float sum = 0.f;
    for (int i = t; i < NN; i += 256) sum += __expf((float)p[i] - mx);
    red[t] = sum; __syncthreads();
    for (int s = 128; s > 0; s >>= 1) {
        if (t < s) red[t] += red[t + s];
        __syncthreads();
    }
    float inv = 1.0f / red[0];
    for (int i = t; i < NN; i += 256)
        p[i] = (_Float16)(__expf((float)p[i] - mx) * inv);
}

// ---------------------------------------------------------------------------
// Kernel 4: softmax of Q over each head's 32 channels, in-place on qt.
// qt layout [b][n][h*32+i] => each thread owns 32 contiguous halves.
// ---------------------------------------------------------------------------
__global__ void softmax_q_kernel(_Float16* __restrict__ qt)
{
    size_t gid = (size_t)blockIdx.x * blockDim.x + threadIdx.x; // B*N*HEADS threads
    _Float16* p = qt + gid * KDIM;
    float v[KDIM];
    float mx = -1e30f;
#pragma unroll
    for (int i = 0; i < KDIM; ++i) { v[i] = (float)p[i]; mx = fmaxf(mx, v[i]); }
    float sum = 0.f;
#pragma unroll
    for (int i = 0; i < KDIM; ++i) { v[i] = __expf(v[i] - mx); sum += v[i]; }
    float inv = 1.0f / sum;
#pragma unroll
    for (int i = 0; i < KDIM; ++i) p[i] = (_Float16)(v[i] * inv);
}

// ---------------------------------------------------------------------------
// Kernel 5a: context partials.  context = key_sm @ V^T per (b,head): [32][64]
// with K=N=4096 split KSPLIT ways for occupancy (grid 64 x 8, 8 waves/block,
// one 16x16 WMMA tile per wave, pipelined K loop).  f32 partials to ctxp.
// ---------------------------------------------------------------------------
__global__ void ctx_part_kernel(const _Float16* __restrict__ kh,
                                const _Float16* __restrict__ vh,
                                float* __restrict__ ctxp)   // [KSPLIT][64][32][64]
{
    int bh = blockIdx.x;                 // 0..63
    int ks = blockIdx.y;                 // 0..KSPLIT-1
    int b = bh >> 3, h = bh & 7;
    int lane = threadIdx.x & 31, wave = threadIdx.x >> 5;
    int m  = wave & 1;                   // kd tile (0..1)
    int nn = wave >> 1;                  // d tile  (0..3)
    int row = lane & 15, half = lane >> 4;
    const _Float16* Ar = kh + ((size_t)b * NHKD + h * KDIM + m * 16 + row) * NN
                            + ks * NSPL + half * 16;
    const _Float16* Br = vh + ((size_t)b * DHC  + h * DD   + nn * 16 + row) * NN
                            + ks * NSPL + half * 16;
    v8f acc = {};
    v16h a0 = load_a16(Ar);
    v16h b0 = *(const v16h*)(Br);
    for (int kc = 0; kc < NSPL - 32; kc += 32) {
        v16h a1 = load_a16(Ar + kc + 32);
        v16h b1 = *(const v16h*)(Br + kc + 32);
        acc = wmma_f16(a0, b0, acc);
        a0 = a1; b0 = b1;
    }
    acc = wmma_f16(a0, b0, acc);
#pragma unroll
    for (int r = 0; r < 8; ++r) {
        int kd = m * 16 + r + half * 8;
        int d  = nn * 16 + row;
        ctxp[(((size_t)ks * 64 + bh) * KDIM + kd) * DD + d] = acc[r];
    }
}

// ---------------------------------------------------------------------------
// Kernel 5b: deterministic reduction of the K-split partials (fixed order)
// + transpose to ctxT[bh][d][kd] f16 (A operand of the attended GEMM).
// ---------------------------------------------------------------------------
__global__ void ctx_reduce_kernel(const float* __restrict__ ctxp,
                                  _Float16* __restrict__ ctxT)
{
    int gid = blockIdx.x * blockDim.x + threadIdx.x;   // 64*32*64 threads
    int bh = gid >> 11;                                // / (32*64)
    int rem = gid & 2047;
    int kd = rem >> 6;
    int d  = rem & 63;
    float s = 0.f;
#pragma unroll
    for (int ks = 0; ks < KSPLIT; ++ks)
        s += ctxp[(((size_t)ks * 64 + bh) * KDIM + kd) * DD + d];
    ctxT[((size_t)bh * DD + d) * KDIM + kd] = (_Float16)s;
}

// ---------------------------------------------------------------------------
// Kernel 6: attended = ctx^T @ query_sm per (b,head): [64][4096], K=32 (one
// WMMA per tile) + fused ReLU.  Output transposed: att_t[b][n][h*64+d].
// ---------------------------------------------------------------------------
__global__ void att_kernel(const _Float16* __restrict__ ctxT,
                           const _Float16* __restrict__ qsm,   // qt, post-softmax
                           _Float16* __restrict__ att_t)
{
    int bh = blockIdx.y;
    int b = bh >> 3, h = bh & 7;
    int lane = threadIdx.x & 31, wave = threadIdx.x >> 5;
    int m  = wave >> 1;                  // d tile (0..3)
    int nw = wave & 1;
    int Nbase = blockIdx.x * 128 + nw * 64;
    int row = lane & 15, half = lane >> 4;
    v16h a = load_a16(ctxT + ((size_t)bh * DD + m * 16 + row) * KDIM + half * 16);
    const _Float16* Bb = qsm + (size_t)b * NN * NHKD;
#pragma unroll
    for (int ns = 0; ns < 4; ++ns) {
        int n = Nbase + ns * 16 + row;
        v16h bv = *(const v16h*)(Bb + (size_t)n * NHKD + h * KDIM + half * 16);
        v8f acc = {};
        acc = wmma_f16(a, bv, acc);
#pragma unroll
        for (int r = 0; r < 8; ++r) {
            int d = m * 16 + r + half * 8;
            float y = acc[r];
            y = y > 0.f ? y : 0.f;       // fused ReLU
            att_t[((size_t)b * NN + n) * DHC + h * DD + d] = (_Float16)y;
        }
    }
}

// ---------------------------------------------------------------------------
// Kernel 7: output projection GEMM + BN affine, f32 result (pipelined).
// A = wph [384][512], B = att_t[b][n][512].
// ---------------------------------------------------------------------------
__global__ void gemm_out_kernel(const _Float16* __restrict__ A,
                                const _Float16* __restrict__ Bt,
                                const float* __restrict__ sp,
                                const float* __restrict__ bp,
                                float* __restrict__ out)     // [B][384][N]
{
    const int K = DHC;
    int b    = blockIdx.z;
    int lane = threadIdx.x & 31;
    int wave = threadIdx.x >> 5;
    int mw = wave >> 1, nw = wave & 1;
    int Mbase = blockIdx.y * 64 + mw * 16;
    int Nbase = blockIdx.x * 128 + nw * 64;
    int row  = lane & 15;
    int half = lane >> 4;
    const _Float16* Ar = A + (size_t)(Mbase + row) * K + half * 16;
    const _Float16* Bb = Bt + ((size_t)b * NN + Nbase) * K + half * 16;

    v8f acc[4] = {};
    v16h a0 = load_a16(Ar);
    v16h b0[4];
#pragma unroll
    for (int ns = 0; ns < 4; ++ns)
        b0[ns] = *(const v16h*)(Bb + (size_t)(ns * 16 + row) * K);
    for (int kc = 0; kc < K - 32; kc += 32) {
        v16h a1 = load_a16(Ar + kc + 32);
        v16h b1[4];
#pragma unroll
        for (int ns = 0; ns < 4; ++ns)
            b1[ns] = *(const v16h*)(Bb + (size_t)(ns * 16 + row) * K + kc + 32);
#pragma unroll
        for (int ns = 0; ns < 4; ++ns)
            acc[ns] = wmma_f16(a0, b0[ns], acc[ns]);
        a0 = a1;
#pragma unroll
        for (int ns = 0; ns < 4; ++ns) b0[ns] = b1[ns];
    }
#pragma unroll
    for (int ns = 0; ns < 4; ++ns)
        acc[ns] = wmma_f16(a0, b0[ns], acc[ns]);

#pragma unroll
    for (int ns = 0; ns < 4; ++ns) {
#pragma unroll
        for (int r = 0; r < 8; ++r) {
            int o = Mbase + r + half * 8;
            int n = Nbase + ns * 16 + row;
            out[((size_t)b * DIMC + o) * NN + n] = acc[ns][r] * sp[o] + bp[o];
        }
    }
}

// ---------------------------------------------------------------------------
// Host-side orchestration
// ---------------------------------------------------------------------------
extern "C" void kernel_launch(void* const* d_in, const int* in_sizes, int n_in,
                              void* d_out, int out_size, void* d_ws, size_t ws_size,
                              hipStream_t stream)
{
    const float* x  = (const float*)d_in[0];
    const float* Wq = (const float*)d_in[1];
    const float* sq = (const float*)d_in[2];
    const float* bq = (const float*)d_in[3];
    const float* Wk = (const float*)d_in[4];
    const float* sk = (const float*)d_in[5];
    const float* bk = (const float*)d_in[6];
    const float* Wv = (const float*)d_in[7];
    const float* sv = (const float*)d_in[8];
    const float* bv = (const float*)d_in[9];
    const float* Wp = (const float*)d_in[10];
    const float* sp = (const float*)d_in[11];
    const float* bp = (const float*)d_in[12];
    float* out = (float*)d_out;

    // Workspace layout (bytes, all 256-aligned)
    char* ws = (char*)d_ws;
    _Float16* xt    = (_Float16*)(ws);                        // 25,165,824
    _Float16* wabc  = (_Float16*)(ws + 25165824);             //    786,432
    _Float16* wph   = (_Float16*)(ws + 25952256);             //    393,216
    float*    sconc = (float*)   (ws + 26345472);             //      4,096
    float*    bconc = (float*)   (ws + 26349568);             //      4,096
    _Float16* qt    = (_Float16*)(ws + 26353664);             // 16,777,216
    _Float16* kh    = (_Float16*)(ws + 43130880);             // 16,777,216
    _Float16* vh    = (_Float16*)(ws + 59908096);             // 33,554,432
    _Float16* ctxT  = (_Float16*)(ws + 93462528);             //    262,144
    _Float16* att_t = (_Float16*)(ws + 93724672);             // 33,554,432
    float*    ctxp  = (float*)   (ws + 127279104);            //  4,194,304
    // total: 131,473,408 bytes

    // 0) weights -> f16 (+ BN param concat)
    prep_weights_kernel<<<1536, 256, 0, stream>>>(
        Wq, sq, bq, Wk, sk, bk, Wv, sv, bv, Wp, wabc, wph, sconc, bconc);

    // 1) x -> xt (f16, n-major)
    xpose_kernel<<<dim3(NN / 32, DIMC / 32, BB), dim3(32, 8), 0, stream>>>(x, xt);

    // 2) QKV projection (WMMA, pipelined) + BN affine
    gemm_qkv_kernel<<<dim3(NN / 128, 1024 / 64, BB), 256, 0, stream>>>(
        wabc, xt, sconc, bconc, qt, kh, vh);

    // 3) softmax K over spatial (2048 rows)
    softmax_k_kernel<<<BB * NHKD, 256, 0, stream>>>(kh);

    // 4) softmax Q over head channels
    softmax_q_kernel<<<(BB * NN * HEADS) / 256, 256, 0, stream>>>(qt);

    // 5) context (WMMA, K-split) + deterministic reduce/transpose
    ctx_part_kernel<<<dim3(BB * HEADS, KSPLIT), 256, 0, stream>>>(kh, vh, ctxp);
    ctx_reduce_kernel<<<(BB * HEADS * KDIM * DD) / 256, 256, 0, stream>>>(ctxp, ctxT);

    // 6) attended (WMMA) + ReLU, stored transposed
    att_kernel<<<dim3(NN / 128, BB * HEADS), 256, 0, stream>>>(ctxT, qt, att_t);

    // 7) output projection (WMMA, pipelined) + BN affine -> f32
    gemm_out_kernel<<<dim3(NN / 128, DIMC / 64, BB), 256, 0, stream>>>(
        wph, att_t, sp, bp, out);

    (void)in_sizes; (void)n_in; (void)out_size; (void)ws_size;
}
